// Attention_40785009443309
// MI455X (gfx1250) — matrix operand order repaired and checked by reading, not compiled, not run on hardware
//
#include <hip/hip_runtime.h>
#include <hip/hip_bf16.h>

// ---------------------------------------------------------------------------
// MHA forward for MI455X (gfx1250, wave32, WMMA + Tensor Data Mover).
// B=4, N=2048, C=1024, H=16, HD=64.
// Pipeline: cvt(f32->bf16) -> QKV GEMM (TDM-staged B, wmma bf16)
//           -> flash attention (TDM-staged K/V, wmma)
//           -> output projection GEMM (TDM-staged B, wmma) + bias -> f32 out.
// ---------------------------------------------------------------------------

typedef __attribute__((ext_vector_type(16))) __bf16 bf16x16;
typedef __attribute__((ext_vector_type(8)))  __bf16 bf16x8;
typedef __attribute__((ext_vector_type(8)))  float  f32x8;
typedef __attribute__((ext_vector_type(4)))  unsigned int u32x4;
typedef __attribute__((ext_vector_type(8)))  int  i32x8;
typedef __attribute__((ext_vector_type(4)))  int  i32x4;

#define DEVINL __device__ __forceinline__

// ----- constants -----------------------------------------------------------
namespace cfg {
constexpr int B  = 4;
constexpr int N  = 2048;
constexpr int C  = 1024;
constexpr int H  = 16;
constexpr int HD = 64;
constexpr int M  = B * N;        // 8192 rows
constexpr int DQKV = 3 * C;      // 3072
constexpr float SCALE = 0.125f;  // HD^-0.5
}

// ----- WMMA wrapper --------------------------------------------------------
DEVINL f32x8 wmma_bf16(bf16x16 a, bf16x16 b, f32x8 c) {
  return __builtin_amdgcn_wmma_f32_16x16x32_bf16(false, a, false, b,
                                                 (short)0, c, false, false);
}

DEVINL f32x8 zero8() {
  f32x8 z = {0.f, 0.f, 0.f, 0.f, 0.f, 0.f, 0.f, 0.f};
  return z;
}

// ----- Tensor Data Mover: 2D tile (bf16) global -> LDS ---------------------
// D# per CDNA5 ISA. tensor dims == tile dims (no OOB).
// tile_d0 = contiguous elements per row, tile_d1 = rows, stride0 in elements.
// LDS result: row-major [tile_d1][tile_d0].
DEVINL unsigned lds_off(const void* p) {
  return (unsigned)(unsigned long long)(uintptr_t)p;  // flat addr low 32 = LDS offset
}

DEVINL void tdm_load_2d_bf16(unsigned ldsoff, const void* gptr,
                             unsigned tile_d0, unsigned tile_d1,
                             unsigned long long stride0) {
  const unsigned long long ga = (unsigned long long)(uintptr_t)gptr;
  u32x4 g0;
  g0[0] = 1u;                                   // count=1 (valid user D#)
  g0[1] = ldsoff;                               // lds_addr (bytes)
  g0[2] = (unsigned)(ga & 0xFFFFFFFFull);       // global_addr[31:0]
  g0[3] = (unsigned)((ga >> 32) & 0x1FFFFFFull) // global_addr[56:32]
        | (2u << 30);                           // type = 2 ("image")
  i32x8 g1;
  g1[0] = (int)(1u << 16);                      // wg_mask=0, data_size=1 (2B)
  g1[1] = (int)((tile_d0 & 0xFFFFu) << 16);     // abar=0 | tensor_dim0[15:0]
  g1[2] = (int)(((tile_d0 >> 16) & 0xFFFFu) | ((tile_d1 & 0xFFFFu) << 16));
  g1[3] = (int)(((tile_d1 >> 16) & 0xFFFFu) | ((tile_d0 & 0xFFFFu) << 16)); // tile_dim0
  g1[4] = (int)(tile_d1 & 0xFFFFu);             // tile_dim1 | tile_dim2=0
  g1[5] = (int)(unsigned)(stride0 & 0xFFFFFFFFull);      // dim0_stride[31:0]
  g1[6] = (int)(unsigned)((stride0 >> 32) & 0xFFFFull);  // [47:32] | dim1_stride lo=0
  g1[7] = 0;
  i32x4 gz4 = {0, 0, 0, 0};
  i32x8 gz8 = {0, 0, 0, 0, 0, 0, 0, 0};
  // clang-23 / therock-10.0 signature: (g0, g1, g2, g3, extra, cpol)
  __builtin_amdgcn_tensor_load_to_lds(g0, g1, gz4, gz4, gz8, 0);
}

// ----- fragment loaders ----------------------------------------------------
// A-fragment (16x32 bf16, M x K), row-major source with leading dim `ld`.
DEVINL bf16x16 load_fragA(const __bf16* __restrict__ base, int ld, int row0,
                          int k0, int lane) {
  const __bf16* p = base + (size_t)(row0 + (lane & 15)) * ld + k0 + ((lane >> 4) << 3);
  bf16x8 lo = *(const bf16x8*)(p);        // K = koff .. koff+7
  bf16x8 hi = *(const bf16x8*)(p + 16);   // K = koff+16 .. koff+23
  bf16x16 r;
#pragma unroll
  for (int i = 0; i < 8; ++i) { r[i] = lo[i]; r[i + 8] = hi[i]; }
  return r;
}

// B-fragment (32x16 bf16, K x N). Column n lives at source row (row0+n),
// K contiguous along that row (works for global or LDS pointers).
DEVINL bf16x16 load_fragB(const __bf16* base, int ld, int row0,
                          int k0, int lane) {
  const __bf16* p = base + (size_t)(row0 + (lane & 15)) * ld + k0 + ((lane >> 4) << 3);
  bf16x8 lo = *(const bf16x8*)(p);
  bf16x8 hi = *(const bf16x8*)(p + 16);
  bf16x16 r;
#pragma unroll
  for (int i = 0; i < 8; ++i) { r[i] = lo[i]; r[i + 8] = hi[i]; }
  return r;
}

// ----- kernel 0: f32 -> bf16 conversion ------------------------------------
__global__ void cvt_f32_bf16(const float* __restrict__ in,
                             __bf16* __restrict__ out, int n) {
  int i = blockIdx.x * blockDim.x + threadIdx.x;
  int stride = gridDim.x * blockDim.x;
  for (; i < n; i += stride) out[i] = (__bf16)in[i];
}

// ----- kernel 1: QKV GEMM ---------------------------------------------------
// qkv[m,d] = sum_k xb[m,k]*wq[d,k]. Block = 4 waves; wave tile 32(M)x64(D);
// block tile 128x64. B (weight) tile TDM-staged to LDS, double-buffered.
// Scatters into Q [BH,N,HD] (scaled), K [BH,N,HD], Vt [BH,HD,N].
__global__ __launch_bounds__(128)
void qkv_gemm(const __bf16* __restrict__ X, const __bf16* __restrict__ W,
              __bf16* __restrict__ Q, __bf16* __restrict__ K,
              __bf16* __restrict__ VT) {
  __shared__ __align__(16) __bf16 bsm[2][64 * 32];  // [d-rows][k], 4 KiB each

  const int lane = threadIdx.x & 31;
  const int wave = threadIdx.x >> 5;
  const int m0 = blockIdx.y * 128 + wave * 32;
  const int n0 = blockIdx.x * 64;

  f32x8 acc[2][4];
#pragma unroll
  for (int mi = 0; mi < 2; ++mi)
#pragma unroll
    for (int nf = 0; nf < 4; ++nf) acc[mi][nf] = zero8();

  constexpr int STEPS = cfg::C / 32;
  if (wave == 0) {
    tdm_load_2d_bf16(lds_off(&bsm[0][0]), W + (size_t)n0 * cfg::C, 32, 64, cfg::C);
    __builtin_amdgcn_s_wait_tensorcnt(0);
  }
  __syncthreads();

  for (int s = 0; s < STEPS; ++s) {
    const int k0 = s * 32;
    const int cur = s & 1;
    if (wave == 0 && s + 1 < STEPS)
      tdm_load_2d_bf16(lds_off(&bsm[1 - cur][0]),
                       W + (size_t)n0 * cfg::C + (k0 + 32), 32, 64, cfg::C);

    if (k0 + 64 < cfg::C)
      __builtin_prefetch(X + (size_t)(m0 + (lane & 15)) * cfg::C + k0 + 64, 0, 0);

    bf16x16 a0 = load_fragA(X, cfg::C, m0, k0, lane);
    bf16x16 a1 = load_fragA(X, cfg::C, m0 + 16, k0, lane);
#pragma unroll
    for (int nf = 0; nf < 4; ++nf) {
      bf16x16 b = load_fragB(&bsm[cur][0], 32, nf * 16, 0, lane);
      acc[0][nf] = wmma_bf16(a0, b, acc[0][nf]);
      acc[1][nf] = wmma_bf16(a1, b, acc[1][nf]);
    }

    if (wave == 0) __builtin_amdgcn_s_wait_tensorcnt(0);
    __syncthreads();
  }

  const int nloc = lane & 15;
#pragma unroll
  for (int mi = 0; mi < 2; ++mi) {
    const int mb = m0 + mi * 16 + ((lane >> 4) << 3);
#pragma unroll
    for (int nf = 0; nf < 4; ++nf) {
      const int d = n0 + nf * 16 + nloc;        // 0..3071
      const int which = d >> 10;                // 0=Q 1=K 2=V
      const int rem = d & 1023;
      const int h = rem >> 6, hd = rem & 63;
#pragma unroll
      for (int v = 0; v < 8; ++v) {
        const int m = mb + v;
        const int bb = m >> 11, n = m & 2047;
        const size_t bh = (size_t)(bb * cfg::H + h);
        const float val = acc[mi][nf][v];
        if (which == 0)
          Q[(bh * cfg::N + n) * cfg::HD + hd] = (__bf16)(val * cfg::SCALE);
        else if (which == 1)
          K[(bh * cfg::N + n) * cfg::HD + hd] = (__bf16)val;
        else
          VT[(bh * cfg::HD + hd) * cfg::N + n] = (__bf16)val;  // transposed
      }
    }
  }
}

// ----- kernel 2: flash attention -------------------------------------------
// Grid (B*H=64, N/128=16); block 256 = 8 waves; each wave owns 16 q rows.
// K/V tiles (shared by all 8 waves) TDM-staged to LDS, double-buffered.
__global__ __launch_bounds__(256)
void attn_fwd(const __bf16* __restrict__ Q, const __bf16* __restrict__ K,
              const __bf16* __restrict__ VT, __bf16* __restrict__ AO) {
  __shared__ __align__(16) __bf16 ksm[2][32 * 64];          // [key][hd] 4 KiB
  __shared__ __align__(16) __bf16 vsm[2][64 * 32];          // [hd][key] 4 KiB
  __shared__ __align__(16) unsigned short pst[8][16 * 32];  // per-wave P tile

  const int lane = threadIdx.x & 31;
  const int wave = threadIdx.x >> 5;
  const int bh = blockIdx.x;
  const int b = bh >> 4, h = bh & 15;

  const __bf16* Qb = Q + (size_t)bh * cfg::N * cfg::HD;
  const __bf16* Kb = K + (size_t)bh * cfg::N * cfg::HD;
  const __bf16* Vb = VT + (size_t)bh * cfg::HD * cfg::N;

  const int q0 = blockIdx.y * 128 + wave * 16;

  const bf16x16 qa0 = load_fragA(Qb, cfg::HD, q0, 0, lane);
  const bf16x16 qa1 = load_fragA(Qb, cfg::HD, q0, 32, lane);

  f32x8 o[4];
#pragma unroll
  for (int c = 0; c < 4; ++c) o[c] = zero8();
  float rm[8], rl[8];
#pragma unroll
  for (int v = 0; v < 8; ++v) { rm[v] = -__builtin_inff(); rl[v] = 0.f; }

  const float L2E = 1.4426950408889634f;
  constexpr int STEPS = cfg::N / 32;

  if (wave == 0) {
    tdm_load_2d_bf16(lds_off(&ksm[0][0]), Kb, 64, 32, cfg::HD);
    tdm_load_2d_bf16(lds_off(&vsm[0][0]), Vb, 32, 64, cfg::N);
    __builtin_amdgcn_s_wait_tensorcnt(0);
  }
  __syncthreads();

  for (int s = 0; s < STEPS; ++s) {
    const int j0 = s * 32;
    const int cur = s & 1;
    if (wave == 0 && s + 1 < STEPS) {
      tdm_load_2d_bf16(lds_off(&ksm[1 - cur][0]),
                       Kb + (size_t)(j0 + 32) * cfg::HD, 64, 32, cfg::HD);
      tdm_load_2d_bf16(lds_off(&vsm[1 - cur][0]),
                       Vb + (j0 + 32), 32, 64, cfg::N);
    }

    // S tile = Q(16x64) @ K^T(64x32), fragments from LDS.
    const __bf16* kt = &ksm[cur][0];
    bf16x16 kb00 = load_fragB(kt, 64, 0, 0, lane);
    bf16x16 kb01 = load_fragB(kt, 64, 0, 32, lane);
    bf16x16 kb10 = load_fragB(kt, 64, 16, 0, lane);
    bf16x16 kb11 = load_fragB(kt, 64, 16, 32, lane);

    f32x8 s0 = wmma_bf16(qa0, kb00, zero8());
    s0 = wmma_bf16(qa1, kb01, s0);
    f32x8 s1 = wmma_bf16(qa0, kb10, zero8());
    s1 = wmma_bf16(qa1, kb11, s1);

    // Online softmax; row m lives in one VGPR across a 16-lane half-wave.
#pragma unroll
    for (int v = 0; v < 8; ++v) {
      float mx = fmaxf(s0[v], s1[v]);
#pragma unroll
      for (int off = 8; off > 0; off >>= 1) mx = fmaxf(mx, __shfl_xor(mx, off, 16));
      const float nm = fmaxf(rm[v], mx);
      const float sc = __builtin_exp2f((rm[v] - nm) * L2E);
      const float p0 = __builtin_exp2f((s0[v] - nm) * L2E);
      const float p1 = __builtin_exp2f((s1[v] - nm) * L2E);
      float sum = p0 + p1;
#pragma unroll
      for (int off = 8; off > 0; off >>= 1) sum += __shfl_xor(sum, off, 16);
      rl[v] = rl[v] * sc + sum;
      rm[v] = nm;
      o[0][v] *= sc; o[1][v] *= sc; o[2][v] *= sc; o[3][v] *= sc;
      s0[v] = p0; s1[v] = p1;
    }

    // Re-layout P (C/D layout) -> A-fragment via per-wave LDS bounce.
    unsigned short* ps = &pst[wave][0];
    {
      const int nloc = lane & 15;
      const int mb = (lane >> 4) << 3;
#pragma unroll
      for (int v = 0; v < 8; ++v) {
        ps[(mb + v) * 32 + nloc]      = __builtin_bit_cast(unsigned short, (__bf16)s0[v]);
        ps[(mb + v) * 32 + 16 + nloc] = __builtin_bit_cast(unsigned short, (__bf16)s1[v]);
      }
    }
    bf16x16 pa;
    {
      const unsigned short* lp = ps + (lane & 15) * 32 + ((lane >> 4) << 3);
      bf16x8 lo = __builtin_bit_cast(bf16x8, *(const u32x4*)lp);
      bf16x8 hi = __builtin_bit_cast(bf16x8, *(const u32x4*)(lp + 16));
#pragma unroll
      for (int i = 0; i < 8; ++i) { pa[i] = lo[i]; pa[i + 8] = hi[i]; }
    }

    // O += P(16x32) @ V(32x64), V fragments from LDS.
#pragma unroll
    for (int c = 0; c < 4; ++c) {
      bf16x16 vb = load_fragB(&vsm[cur][0], 32, c * 16, 0, lane);
      o[c] = wmma_bf16(pa, vb, o[c]);
    }

    if (wave == 0) __builtin_amdgcn_s_wait_tensorcnt(0);
    __syncthreads();
  }

  // Finalize: O /= l, store bf16 into AO [B*N, C].
  const int nloc = lane & 15;
  const int mb = (lane >> 4) << 3;
  float inv[8];
#pragma unroll
  for (int v = 0; v < 8; ++v) inv[v] = rl[v] > 0.f ? 1.f / rl[v] : 0.f;
#pragma unroll
  for (int c = 0; c < 4; ++c) {
    const int col = h * cfg::HD + c * 16 + nloc;
#pragma unroll
    for (int v = 0; v < 8; ++v) {
      const size_t row = (size_t)b * cfg::N + q0 + mb + v;
      AO[row * cfg::C + col] = (__bf16)(o[c][v] * inv[v]);
    }
  }
}

// ----- kernel 3: output projection GEMM ------------------------------------
// out[m,d] = sum_k ao[m,k]*wp[d,k] + bias[d]; M=8192, N=1024, K=1024.
__global__ __launch_bounds__(128)
void proj_gemm(const __bf16* __restrict__ AO, const __bf16* __restrict__ W,
               const float* __restrict__ bias, float* __restrict__ out) {
  __shared__ __align__(16) __bf16 bsm[2][64 * 32];

  const int lane = threadIdx.x & 31;
  const int wave = threadIdx.x >> 5;
  const int m0 = blockIdx.y * 128 + wave * 32;
  const int n0 = blockIdx.x * 64;

  f32x8 acc[2][4];
#pragma unroll
  for (int mi = 0; mi < 2; ++mi)
#pragma unroll
    for (int nf = 0; nf < 4; ++nf) acc[mi][nf] = zero8();

  constexpr int STEPS = cfg::C / 32;
  if (wave == 0) {
    tdm_load_2d_bf16(lds_off(&bsm[0][0]), W + (size_t)n0 * cfg::C, 32, 64, cfg::C);
    __builtin_amdgcn_s_wait_tensorcnt(0);
  }
  __syncthreads();

  for (int s = 0; s < STEPS; ++s) {
    const int k0 = s * 32;
    const int cur = s & 1;
    if (wave == 0 && s + 1 < STEPS)
      tdm_load_2d_bf16(lds_off(&bsm[1 - cur][0]),
                       W + (size_t)n0 * cfg::C + (k0 + 32), 32, 64, cfg::C);

    if (k0 + 64 < cfg::C)
      __builtin_prefetch(AO + (size_t)(m0 + (lane & 15)) * cfg::C + k0 + 64, 0, 0);

    bf16x16 a0 = load_fragA(AO, cfg::C, m0, k0, lane);
    bf16x16 a1 = load_fragA(AO, cfg::C, m0 + 16, k0, lane);
#pragma unroll
    for (int nf = 0; nf < 4; ++nf) {
      bf16x16 b = load_fragB(&bsm[cur][0], 32, nf * 16, 0, lane);
      acc[0][nf] = wmma_bf16(a0, b, acc[0][nf]);
      acc[1][nf] = wmma_bf16(a1, b, acc[1][nf]);
    }

    if (wave == 0) __builtin_amdgcn_s_wait_tensorcnt(0);
    __syncthreads();
  }

  const int nloc = lane & 15;
#pragma unroll
  for (int mi = 0; mi < 2; ++mi) {
    const int mb = m0 + mi * 16 + ((lane >> 4) << 3);
#pragma unroll
    for (int nf = 0; nf < 4; ++nf) {
      const int d = n0 + nf * 16 + nloc;
      const float bv = bias[d];
#pragma unroll
      for (int v = 0; v < 8; ++v) {
        const size_t m = (size_t)(mb + v);
        out[m * cfg::C + d] = acc[mi][nf][v] + bv;
      }
    }
  }
}

// ----- launch ---------------------------------------------------------------
extern "C" void kernel_launch(void* const* d_in, const int* in_sizes, int n_in,
                              void* d_out, int out_size, void* d_ws, size_t ws_size,
                              hipStream_t stream) {
  (void)in_sizes; (void)n_in; (void)out_size; (void)ws_size;

  const float* x      = (const float*)d_in[0];  // [B,N,C]
  const float* qkv_w  = (const float*)d_in[1];  // [3C,C]
  const float* proj_w = (const float*)d_in[2];  // [C,C]
  const float* proj_b = (const float*)d_in[3];  // [C]
  float* out = (float*)d_out;

  char* ws = (char*)d_ws;
  __bf16* xb = (__bf16*)(ws + 0);          // 16 MiB
  __bf16* wq = (__bf16*)(ws + 16777216);   //  6 MiB
  __bf16* wp = (__bf16*)(ws + 23068672);   //  2 MiB
  __bf16* q  = (__bf16*)(ws + 25165824);   // [BH,N,HD] 16 MiB
  __bf16* k  = (__bf16*)(ws + 41943040);   // [BH,N,HD] 16 MiB
  __bf16* vt = (__bf16*)(ws + 58720256);   // [BH,HD,N] 16 MiB
  __bf16* ao = (__bf16*)(ws + 75497472);   // [B*N,C]   16 MiB

  const int nx  = cfg::M * cfg::C;
  const int nwq = cfg::DQKV * cfg::C;
  const int nwp = cfg::C * cfg::C;

  cvt_f32_bf16<<<4096, 256, 0, stream>>>(x, xb, nx);
  cvt_f32_bf16<<<4096, 256, 0, stream>>>(qkv_w, wq, nwq);
  cvt_f32_bf16<<<4096, 256, 0, stream>>>(proj_w, wp, nwp);

  qkv_gemm<<<dim3(cfg::DQKV / 64, cfg::M / 128), 128, 0, stream>>>(xb, wq, q, k, vt);
  attn_fwd<<<dim3(cfg::B * cfg::H, cfg::N / 128), 256, 0, stream>>>(q, k, vt, ao);
  proj_gemm<<<dim3(cfg::C / 64, cfg::M / 128), 128, 0, stream>>>(ao, wp, proj_b, out);
}
